// MoE_Adapter_46385646796969
// MI455X (gfx1250) — matrix operand order repaired and checked by reading, not compiled, not run on hardware
//
#include <hip/hip_runtime.h>
#include <hip/hip_bf16.h>

// ---------------------------------------------------------------------------
// MoE adapter for MI455X (gfx1250, wave32, WMMA + Tensor Data Mover).
//   out = x@base_W^T + b  +  (x@shared_A^T)@shared_B^T  +  gated routed experts
// Fused as:  H = x@Acat^T ; G = gate(H) ;  out = x@base_W^T + G@Bcat^T + b
// bf16 operands, fp32 WMMA accumulation (v_wmma_f32_16x16x32_bf16).
// Tile staging via TDM (tensor_load_to_lds, TENSORcnt), 3-stage LDS pipeline.
// ---------------------------------------------------------------------------

typedef __attribute__((ext_vector_type(16))) __bf16 v16bf;
typedef __attribute__((ext_vector_type(8)))  __bf16 v8bf;
typedef __attribute__((ext_vector_type(8)))  float  v8f;
typedef __attribute__((ext_vector_type(4)))  unsigned int u32x4;
typedef __attribute__((ext_vector_type(8)))  unsigned int u32x8;

#define TOK    32768         // B*S tokens
#define DDIM   1024          // model dim
#define NEXP   6             // experts
#define RANK   16
#define CDIM   112           // RANK + NEXP*RANK
#define CPAD   128           // CDIM padded to WMMA K multiple
#define LDK    32            // K per LDS stage (one wmma K)
#define LSTR   40            // LDS row stride in bf16 (64B row + 16B TDM pad)
#define LSTAGE 3             // TDM pipeline depth

// ---------------------------------------------------------------------------
// TDM: DMA one 2D bf16 tile (tile_rows x 32 elems) global -> LDS.
// LDS padding: 4 DWORDs after every 16 DWORDs => 80B LDS row stride (= LSTR).
// Issue from one wave only; tracked with TENSORcnt.
// ---------------------------------------------------------------------------
__device__ __forceinline__ void tdm_load_2d(
    unsigned lds_addr, const void* gbase, long long byte_off,
    unsigned tensor_d0, unsigned tensor_rows, unsigned tile_rows,
    unsigned stride_elems)
{
    unsigned long long ga = (unsigned long long)gbase + (unsigned long long)byte_off;
    u32x4 g0;
    g0.x = 1u;                                            // count=1, user D#
    g0.y = lds_addr;                                      // LDS byte address
    g0.z = (unsigned)(ga & 0xFFFFFFFFu);                  // global_addr[31:0]
    g0.w = (unsigned)((ga >> 32) & 0x01FFFFFFu)           // global_addr[56:32]
         | (2u << 30);                                    // type=2 (image)
    u32x8 g1;
    g1.s0 = (1u << 16)                                    // data_size = 2 bytes
          | (1u << 20)                                    // pad_enable
          | (3u << 22)                                    // pad_interval: 16 DWORDs
          | (3u << 25);                                   // pad_amount: 4 DWORDs
    g1.s1 = (tensor_d0 & 0xFFFFu) << 16;                  // tensor_dim0[15:0] @bits63:48
    g1.s2 = ((tensor_d0 >> 16) & 0xFFFFu)                 // tensor_dim0[31:16]
          | ((tensor_rows & 0xFFFFu) << 16);              // tensor_dim1[15:0]
    g1.s3 = ((tensor_rows >> 16) & 0xFFFFu)               // tensor_dim1[31:16]
          | ((unsigned)LDK << 16);                        // tile_dim0 = 32 elems
    g1.s4 = tile_rows & 0xFFFFu;                          // tile_dim1 (tile_dim2=0)
    g1.s5 = stride_elems;                                 // tensor_dim0_stride[31:0]
    g1.s6 = 0u;                                           // stride hi / dim1_stride
    g1.s7 = 0u;
    asm volatile("tensor_load_to_lds %0, %1" :: "s"(g0), "s"(g1) : "memory");
}

// ---------------------------------------------------------------------------
// P1: cast base_W -> bf16 [o,k]; build Acat [112,1024] and Bcat [1024,128] bf16
// ---------------------------------------------------------------------------
__global__ __launch_bounds__(256) void moe_prep_weights(
    const float* __restrict__ base_W, const float* __restrict__ shared_A,
    const float* __restrict__ shared_B, const float* __restrict__ routed_A,
    const float* __restrict__ routed_B,
    __bf16* __restrict__ wb, __bf16* __restrict__ acat, __bf16* __restrict__ bcat)
{
    const int stride = gridDim.x * blockDim.x;
    const int tid0 = blockIdx.x * blockDim.x + threadIdx.x;

    for (int i = tid0; i < DDIM * DDIM; i += stride)           // base_W, keep [o,k]
        wb[i] = (__bf16)base_W[i];

    for (int i = tid0; i < CDIM * DDIM; i += stride)           // Acat = shared_A ++ routed_A
        acat[i] = (__bf16)((i < RANK * DDIM) ? shared_A[i] : routed_A[i - RANK * DDIM]);

    for (int i = tid0; i < DDIM * CPAD; i += stride) {         // Bcat[o][c], zero pad c>=112
        int o = i >> 7, c = i & (CPAD - 1);
        float v = 0.f;
        if (c < RANK)      v = shared_B[o * RANK + c];
        else if (c < CDIM) {
            int e = (c - RANK) >> 4, r = (c - RANK) & 15;
            v = routed_B[((size_t)e * DDIM + o) * RANK + r];
        }
        bcat[i] = (__bf16)v;
    }
}

// ---------------------------------------------------------------------------
// P2: one wave per token: cast x->bf16, router logits/softmax/top-2/renorm
// ---------------------------------------------------------------------------
__global__ __launch_bounds__(256) void moe_router_cast(
    const float* __restrict__ x, const float* __restrict__ rW,
    const float* __restrict__ rb, __bf16* __restrict__ xb,
    float* __restrict__ wgate)
{
    __shared__ float sRW[NEXP * DDIM];
    for (int i = threadIdx.x; i < NEXP * DDIM; i += 256) sRW[i] = rW[i];
    __syncthreads();

    const int lane = threadIdx.x & 31;
    const int wv   = threadIdx.x >> 5;
    const int n    = blockIdx.x * 8 + wv;

    const float* xr = x + (size_t)n * DDIM;
    __bf16*      xo = xb + (size_t)n * DDIM;

    float acc[NEXP] = {0.f, 0.f, 0.f, 0.f, 0.f, 0.f};
    for (int i = 0; i < DDIM / 32; ++i) {
        int d = i * 32 + lane;
        float v = xr[d];
        xo[d] = (__bf16)v;
#pragma unroll
        for (int e = 0; e < NEXP; ++e) acc[e] += v * sRW[e * DDIM + d];
    }
#pragma unroll
    for (int off = 16; off > 0; off >>= 1)
#pragma unroll
        for (int e = 0; e < NEXP; ++e) acc[e] += __shfl_xor(acc[e], off, 32);

    if (lane == 0) {
        float p[NEXP], mx = -1e30f;
#pragma unroll
        for (int e = 0; e < NEXP; ++e) { p[e] = acc[e] + rb[e]; mx = fmaxf(mx, p[e]); }
        float s = 0.f;
#pragma unroll
        for (int e = 0; e < NEXP; ++e) { p[e] = __expf(p[e] - mx); s += p[e]; }
        float inv = 1.f / s;
#pragma unroll
        for (int e = 0; e < NEXP; ++e) p[e] *= inv;
        int i1 = 0;
#pragma unroll
        for (int e = 1; e < NEXP; ++e) if (p[e] > p[i1]) i1 = e;
        int i2 = (i1 == 0) ? 1 : 0;
#pragma unroll
        for (int e = 0; e < NEXP; ++e) if (e != i1 && p[e] > p[i2]) i2 = e;
        float denom = 1.f / (p[i1] + p[i2] + 1e-6f);
#pragma unroll
        for (int e = 0; e < NEXP; ++e)
            wgate[(size_t)n * NEXP + e] = (e == i1 || e == i2) ? p[e] * denom : 0.f;
    }
}

// ---------------------------------------------------------------------------
// G1: H = x_bf16 @ Acat^T  [N x 112], gate columns, emit G bf16 [N x 128]
//     128 tokens/block, 8 waves; TDM-staged 3-deep LDS pipeline.
// ---------------------------------------------------------------------------
__global__ __launch_bounds__(256) void moe_lowrank_gate(
    const __bf16* __restrict__ xb, const __bf16* __restrict__ acat,
    const float* __restrict__ wgate, __bf16* __restrict__ gmat)
{
    __shared__ __bf16 sA[LSTAGE][128 * LSTR];
    __shared__ __bf16 sB[LSTAGE][CDIM * LSTR];

    const int tid = threadIdx.x, lane = tid & 31, wv = tid >> 5;
    const int h = lane >> 4, lm = lane & 15;
    const int rowBase = blockIdx.x * 128;

    const unsigned sAb = (unsigned)(uintptr_t)&sA[0][0];
    const unsigned sBb = (unsigned)(uintptr_t)&sB[0][0];
    const unsigned aBytes = 128 * LSTR * 2, bBytes = CDIM * LSTR * 2;

    auto issue = [&](int s) {
        const int st = s % LSTAGE;
        const long long kb = (long long)s * LDK * 2;
        tdm_load_2d(sAb + st * aBytes, xb, (long long)rowBase * DDIM * 2 + kb,
                    DDIM, TOK, 128, DDIM);
        tdm_load_2d(sBb + st * bBytes, acat, kb, DDIM, CDIM, CDIM, DDIM);
    };

    v8f acc[7];
#pragma unroll
    for (int nt = 0; nt < 7; ++nt) acc[nt] = (v8f){0.f,0.f,0.f,0.f,0.f,0.f,0.f,0.f};

    const int NSTEP = DDIM / LDK;   // 32
    if (wv == 0) { issue(0); issue(1); issue(2); }
    for (int s = 0; s < NSTEP; ++s) {
        if (wv == 0) __builtin_amdgcn_s_wait_tensorcnt((short)(2 * (LSTAGE - 1)));
        __syncthreads();
        const int st = s % LSTAGE;
        const __bf16* pa = &sA[st][(wv * 16 + lm) * LSTR + h * 16];
        v8bf alo = *(const v8bf*)(pa);
        v8bf ahi = *(const v8bf*)(pa + 8);
        v16bf afr = __builtin_shufflevector(alo, ahi, 0,1,2,3,4,5,6,7,8,9,10,11,12,13,14,15);
#pragma unroll
        for (int nt = 0; nt < 7; ++nt) {
            const __bf16* pb = &sB[st][(nt * 16 + lm) * LSTR + h * 16];
            v8bf blo = *(const v8bf*)(pb);
            v8bf bhi = *(const v8bf*)(pb + 8);
            v16bf bfr = __builtin_shufflevector(blo, bhi, 0,1,2,3,4,5,6,7,8,9,10,11,12,13,14,15);
            acc[nt] = __builtin_amdgcn_wmma_f32_16x16x32_bf16(
                false, afr, false, bfr, (short)0, acc[nt], false, false);
        }
        __syncthreads();
        if (wv == 0 && s + LSTAGE < NSTEP) issue(s + LSTAGE);
    }

    // epilogue: apply gate weights, write bf16 G, zero pad cols [112,128)
#pragma unroll
    for (int nt = 0; nt < 7; ++nt) {
        const int col = nt * 16 + lm;
#pragma unroll
        for (int j = 0; j < 8; ++j) {
            const int tokn = rowBase + wv * 16 + j + 8 * h;
            float wgt = 1.f;
            if (col >= RANK) wgt = wgate[(size_t)tokn * NEXP + ((col - RANK) >> 4)];
            gmat[(size_t)tokn * CPAD + col] = (__bf16)(acc[nt][j] * wgt);
        }
    }
#pragma unroll
    for (int j = 0; j < 8; ++j) {
        const int tokn = rowBase + wv * 16 + j + 8 * h;
        gmat[(size_t)tokn * CPAD + CDIM + lm] = (__bf16)0.f;
    }
}

// ---------------------------------------------------------------------------
// G2: out = x_bf16 @ base_W^T  +  G @ Bcat^T  +  bias   (fp32 out)
//     128x128 block, 8 waves (2x4), wave tile 64x32, 36 K-steps of 32.
//     TDM-staged 3-deep LDS pipeline (60KB LDS).
// ---------------------------------------------------------------------------
__global__ __launch_bounds__(256) void moe_main_gemm(
    const __bf16* __restrict__ xb, const __bf16* __restrict__ wb,
    const __bf16* __restrict__ gmat, const __bf16* __restrict__ bcat,
    const float* __restrict__ bias, float* __restrict__ out)
{
    __shared__ __bf16 sA[LSTAGE][128 * LSTR];
    __shared__ __bf16 sB[LSTAGE][128 * LSTR];
    __shared__ float  sBias[128];

    const int tid = threadIdx.x, lane = tid & 31, wv = tid >> 5;
    const int wm = wv >> 2, wn = wv & 3;
    const int h = lane >> 4, lm = lane & 15;
    const int bm = (int)blockIdx.x >> 3, bn = (int)blockIdx.x & 7;
    const int rowBase = bm * 128, colBase = bn * 128;

    if (tid < 128) sBias[tid] = bias[colBase + tid];

    const unsigned sAb = (unsigned)(uintptr_t)&sA[0][0];
    const unsigned sBb = (unsigned)(uintptr_t)&sB[0][0];
    const unsigned stBytes = 128 * LSTR * 2;

    auto issue = [&](int s) {
        const int st = s % LSTAGE;
        if (s < DDIM / LDK) {
            const long long kb = (long long)s * LDK * 2;
            tdm_load_2d(sAb + st * stBytes, xb, (long long)rowBase * DDIM * 2 + kb,
                        DDIM, TOK, 128, DDIM);
            tdm_load_2d(sBb + st * stBytes, wb, (long long)colBase * DDIM * 2 + kb,
                        DDIM, DDIM, 128, DDIM);
        } else {
            const long long kb = (long long)(s - DDIM / LDK) * LDK * 2;
            tdm_load_2d(sAb + st * stBytes, gmat, (long long)rowBase * CPAD * 2 + kb,
                        CPAD, TOK, 128, CPAD);
            tdm_load_2d(sBb + st * stBytes, bcat, (long long)colBase * CPAD * 2 + kb,
                        CPAD, DDIM, 128, CPAD);
        }
    };

    v8f acc[4][2];
#pragma unroll
    for (int mt = 0; mt < 4; ++mt)
#pragma unroll
        for (int nt = 0; nt < 2; ++nt) acc[mt][nt] = (v8f){0.f,0.f,0.f,0.f,0.f,0.f,0.f,0.f};

    const int NSTEP = DDIM / LDK + CPAD / LDK;   // 32 + 4
    if (wv == 0) { issue(0); issue(1); issue(2); }
    for (int s = 0; s < NSTEP; ++s) {
        if (wv == 0) __builtin_amdgcn_s_wait_tensorcnt((short)(2 * (LSTAGE - 1)));
        __syncthreads();
        const int st = s % LSTAGE;
        v16bf bfr[2];
#pragma unroll
        for (int nt = 0; nt < 2; ++nt) {
            const __bf16* pb = &sB[st][(wn * 32 + nt * 16 + lm) * LSTR + h * 16];
            v8bf blo = *(const v8bf*)(pb);
            v8bf bhi = *(const v8bf*)(pb + 8);
            bfr[nt] = __builtin_shufflevector(blo, bhi, 0,1,2,3,4,5,6,7,8,9,10,11,12,13,14,15);
        }
#pragma unroll
        for (int mt = 0; mt < 4; ++mt) {
            const __bf16* pa = &sA[st][(wm * 64 + mt * 16 + lm) * LSTR + h * 16];
            v8bf alo = *(const v8bf*)(pa);
            v8bf ahi = *(const v8bf*)(pa + 8);
            v16bf afr = __builtin_shufflevector(alo, ahi, 0,1,2,3,4,5,6,7,8,9,10,11,12,13,14,15);
#pragma unroll
            for (int nt = 0; nt < 2; ++nt)
                acc[mt][nt] = __builtin_amdgcn_wmma_f32_16x16x32_bf16(
                    false, afr, false, bfr[nt], (short)0, acc[mt][nt], false, false);
        }
        __syncthreads();
        if (wv == 0 && s + LSTAGE < NSTEP) issue(s + LSTAGE);
    }

#pragma unroll
    for (int mt = 0; mt < 4; ++mt)
#pragma unroll
        for (int nt = 0; nt < 2; ++nt) {
            const int col = wn * 32 + nt * 16 + lm;
#pragma unroll
            for (int j = 0; j < 8; ++j) {
                const int row = rowBase + wm * 64 + mt * 16 + j + 8 * h;
                out[(size_t)row * DDIM + colBase + col] = acc[mt][nt][j] + sBias[col];
            }
        }
}

// ---------------------------------------------------------------------------
extern "C" void kernel_launch(void* const* d_in, const int* in_sizes, int n_in,
                              void* d_out, int out_size, void* d_ws, size_t ws_size,
                              hipStream_t stream)
{
    const float* x        = (const float*)d_in[0];
    const float* base_W   = (const float*)d_in[1];
    const float* base_b   = (const float*)d_in[2];
    const float* shared_A = (const float*)d_in[3];
    const float* shared_B = (const float*)d_in[4];
    const float* routed_A = (const float*)d_in[5];
    const float* routed_B = (const float*)d_in[6];
    const float* router_W = (const float*)d_in[7];
    const float* router_b = (const float*)d_in[8];
    float* out = (float*)d_out;

    char* ws = (char*)d_ws;
    size_t o = 0;
    __bf16* xb    = (__bf16*)(ws + o); o += (size_t)TOK  * DDIM * 2;   // 64 MB
    __bf16* wb    = (__bf16*)(ws + o); o += (size_t)DDIM * DDIM * 2;   // 2 MB
    __bf16* acat  = (__bf16*)(ws + o); o += (size_t)CDIM * DDIM * 2;
    __bf16* bcat  = (__bf16*)(ws + o); o += (size_t)DDIM * CPAD * 2;
    __bf16* gmat  = (__bf16*)(ws + o); o += (size_t)TOK  * CPAD * 2;   // 8 MB
    float*  wgate = (float*) (ws + o); o += (size_t)TOK  * NEXP * 4;

    moe_prep_weights<<<1024, 256, 0, stream>>>(base_W, shared_A, shared_B,
                                               routed_A, routed_B, wb, acat, bcat);
    moe_router_cast<<<TOK / 8, 256, 0, stream>>>(x, router_W, router_b, xb, wgate);
    moe_lowrank_gate<<<TOK / 128, 256, 0, stream>>>(xb, acat, wgate, gmat);
    moe_main_gemm<<<(TOK / 128) * (DDIM / 128), 256, 0, stream>>>(xb, wb, gmat, bcat,
                                                                  base_b, out);
}
